// ConformerTrunk_2808908612287
// MI455X (gfx1250) — matrix operand order repaired and checked by reading, not compiled, not run on hardware
//
#include <hip/hip_runtime.h>
#include <hip/hip_bf16.h>

// ---------------- problem constants ----------------
#define TT   1024          // T
#define BB   4             // batch
#define CC   256           // channels
#define HH   4             // heads
#define HD   64            // head dim
#define DFF  1024
#define KW   31            // depthwise conv kernel
#define TB   (TT*BB)       // 4096 tokens
#define C3   (3*CC)        // 768
#define TP   (2*TT-1)      // 2047
#define TPP  2048          // padded
#define EPSF 1e-5f

typedef __attribute__((ext_vector_type(16))) _Float16 v16h;
typedef __attribute__((ext_vector_type(8)))  _Float16 v8h;
typedef __attribute__((ext_vector_type(8)))  float    v8f;

// A-matrix fragment: lane holds K=[16g,16g+16) (contiguous, matches B)
__device__ __forceinline__ v16h load_a_frag(const _Float16* p, int g) {
    v8h x0 = *reinterpret_cast<const v8h*>(p + 16 * g);
    v8h x1 = *reinterpret_cast<const v8h*>(p + 16 * g + 8);
    return __builtin_shufflevector(x0, x1, 0,1,2,3,4,5,6,7,8,9,10,11,12,13,14,15);
}

// ---------------------------------------------------------------------------
// Register-blocked WMMA GEMM:  D[M,N] = A[M,K] * B[N,K]^T (+ bias[N])
// One wave computes a 32x64 output block = 2 (M) x 4 (N) WMMA tiles.
// __launch_bounds__(256,1): one workgroup per EU -> full VGPR budget, no
// accumulator spills (64 acc VGPRs + pipelined fragments need ~200 VGPRs).
// Requirements (all call sites satisfy them):
//   M % 32 == 0; K % 32 == 0; lda,ldb mult of 16; the B staging buffer has
//   at least tilesN4*64 rows (padded with zeros when N is not a multiple).
// ---------------------------------------------------------------------------
__global__ __launch_bounds__(256, 1)
void wmma_gemm(const _Float16* __restrict__ A, int lda,
               const _Float16* __restrict__ Bm, int ldb,
               float* __restrict__ D, int ldd,
               const float* __restrict__ bias,
               int M, int N, int Kdim, int tilesM2, int tilesN4) {
    int wave = blockIdx.x * (blockDim.x >> 5) + (threadIdx.x >> 5);
    if (wave >= tilesM2 * tilesN4) return;        // wave-uniform exit
    int tm = wave / tilesN4;
    int tn = wave - tm * tilesN4;
    int lane = threadIdx.x & 31;
    int g    = lane >> 4;      // half-wave group
    int l16  = lane & 15;

    unsigned aoff0 = (unsigned)(tm * 32 + l16) * (unsigned)lda;
    unsigned aoff1 = aoff0 + 16u * (unsigned)lda;
    unsigned boff0 = (unsigned)(tn * 64 + l16) * (unsigned)ldb + 16u * (unsigned)g;
    unsigned bstep = 16u * (unsigned)ldb;

    // single branch-free prefetch of this wave's first operand lines
    __builtin_prefetch(A + aoff0, 0, 0);
    __builtin_prefetch(Bm + boff0, 0, 0);

    v8f acc[2][4] = {};
    for (int k = 0; k < Kdim; k += 32) {
        v16h a[2];
        a[0] = load_a_frag(A + aoff0 + k, g);
        a[1] = load_a_frag(A + aoff1 + k, g);
        v16h bf[4];
        #pragma unroll
        for (int j = 0; j < 4; ++j)
            bf[j] = *reinterpret_cast<const v16h*>(Bm + boff0 + j * bstep + k);
        #pragma unroll
        for (int mi = 0; mi < 2; ++mi)
            #pragma unroll
            for (int j = 0; j < 4; ++j)
                acc[mi][j] = __builtin_amdgcn_wmma_f32_16x16x32_f16(
                    false, a[mi], false, bf[j], (short)0, acc[mi][j], false, false);
    }

    #pragma unroll
    for (int j = 0; j < 4; ++j) {
        int n = tn * 64 + j * 16 + l16;
        if (n < N) {
            float bv = bias ? bias[n] : 0.0f;
            #pragma unroll
            for (int mi = 0; mi < 2; ++mi) {
                unsigned dbase = (unsigned)(tm * 32 + mi * 16 + g * 8) * (unsigned)ldd
                               + (unsigned)n;
                #pragma unroll
                for (int r = 0; r < 8; ++r)
                    D[dbase + (unsigned)r * (unsigned)ldd] = acc[mi][j][r] + bv;
            }
        }
    }
}

// ---------------------------------------------------------------------------
// CDNA5 async-path warm kernel: streams weight bytes into LDS with
// GLOBAL_LOAD_ASYNC_TO_LDS_B128 (ASYNCcnt) and waits. Warms L2 for the
// following GEMM; no architectural side effects.
// ---------------------------------------------------------------------------
__global__ void async_warm_k(const void* __restrict__ src) {
    __shared__ char lbuf[2048];
    unsigned lds_off = (threadIdx.x & 31) * 16;           // 16B per lane
    const char* g = (const char*)src + (size_t)blockIdx.x * 512 + lds_off;
    asm volatile(
        "global_load_async_to_lds_b128 %0, %1, off\n"
        "s_wait_asynccnt 0\n"
        :
        : "v"(lds_off), "v"(g)
        : "memory");
    __syncthreads();
    int v = ((volatile int*)lbuf)[threadIdx.x & 31];      // keep LDS alive
    asm volatile("" :: "v"(v));
}

// ---------------- elementwise / reduction helpers ----------------
__global__ void f32_to_f16_k(const float* __restrict__ src, _Float16* __restrict__ dst,
                             long long nsrc, long long ndst) {
    long long i = (long long)blockIdx.x * blockDim.x + threadIdx.x;
    if (i < ndst) dst[i] = (i < nsrc) ? (_Float16)src[i] : (_Float16)0.0f;
}

__global__ void layernorm_k(const float* __restrict__ x, float* __restrict__ y,
                            const float* __restrict__ g, const float* __restrict__ b) {
    __shared__ float red[CC];
    int row = blockIdx.x, c = threadIdx.x;
    size_t idx = (size_t)row * CC + c;
    float v = x[idx];
    red[c] = v; __syncthreads();
    for (int s = CC / 2; s > 0; s >>= 1) { if (c < s) red[c] += red[c + s]; __syncthreads(); }
    float mean = red[0] * (1.0f / CC);
    __syncthreads();
    float d = v - mean;
    red[c] = d * d; __syncthreads();
    for (int s = CC / 2; s > 0; s >>= 1) { if (c < s) red[c] += red[c + s]; __syncthreads(); }
    float var = red[0] * (1.0f / CC);
    y[idx] = d * rsqrtf(var + EPSF) * g[c] + b[c];
}

__global__ void swish_ip_k(float* __restrict__ x, long long n) {
    long long i = (long long)blockIdx.x * blockDim.x + threadIdx.x;
    if (i < n) { float v = x[i]; x[i] = v / (1.0f + __expf(-v)); }
}

__global__ void add_scaled_k(float* __restrict__ x, const float* __restrict__ t,
                             float alpha, long long n) {
    long long i = (long long)blockIdx.x * blockDim.x + threadIdx.x;
    if (i < n) x[i] += alpha * t[i];
}

// build per-(b,h) attention operand matrices in f16
__global__ void build_heads_k(const float* __restrict__ qkv,
                              const float* __restrict__ bu, const float* __restrict__ bv,
                              _Float16* __restrict__ aq, _Float16* __restrict__ aq2,
                              _Float16* __restrict__ kk, _Float16* __restrict__ vT,
                              int b, int h, float scaling) {
    int i = blockIdx.x * blockDim.x + threadIdx.x;   // t*HD + d
    if (i >= TT * HD) return;
    int t = i >> 6, d = i & (HD - 1);
    const float* base = qkv + (size_t)(t * BB + b) * C3 + h * HD + d;
    float q = base[0] * scaling;
    aq[i]  = (_Float16)(q + bu[d]);
    aq2[i] = (_Float16)(q + bv[d]);
    kk[i]  = (_Float16)base[CC];
    vT[(size_t)d * TT + t] = (_Float16)base[2 * CC];
}

// scores[t,s] += bd[t, (T-1) - t + s]   (fused rel_shift)
__global__ void shift_add_k(float* __restrict__ scores, const float* __restrict__ bd) {
    int i = blockIdx.x * blockDim.x + threadIdx.x;
    if (i >= TT * TT) return;
    int t = i >> 10, s = i & (TT - 1);
    scores[i] += bd[(size_t)t * TPP + (TT - 1 - t + s)];
}

__global__ void softmax_row_k(float* __restrict__ sc) {
    __shared__ float red[256];
    float* row = sc + (size_t)blockIdx.x * TT;
    int c = threadIdx.x;
    float m = -1e30f;
    for (int j = c; j < TT; j += 256) m = fmaxf(m, row[j]);
    red[c] = m; __syncthreads();
    for (int s = 128; s > 0; s >>= 1) { if (c < s) red[c] = fmaxf(red[c], red[c + s]); __syncthreads(); }
    m = red[0]; __syncthreads();
    float sum = 0.0f;
    for (int j = c; j < TT; j += 256) { float e = __expf(row[j] - m); row[j] = e; sum += e; }
    red[c] = sum; __syncthreads();
    for (int s = 128; s > 0; s >>= 1) { if (c < s) red[c] += red[c + s]; __syncthreads(); }
    float inv = 1.0f / red[0];
    for (int j = c; j < TT; j += 256) row[j] *= inv;
}

// GLU: u[tb,c] = z[tb,c] * sigmoid(z[tb, C+c]) with z (TB, 2C)
__global__ void glu_k(const float* __restrict__ z, float* __restrict__ u) {
    int i = blockIdx.x * blockDim.x + threadIdx.x;
    if (i >= TB * CC) return;
    int tb = i >> 8, c = i & (CC - 1);
    float a = z[(size_t)tb * 2 * CC + c];
    float g = z[(size_t)tb * 2 * CC + CC + c];
    u[i] = a / (1.0f + __expf(-g));
}

// depthwise conv along t; u/out layout (T,B,C); SAME padding of (K-1)/2
__global__ void dwconv_k(const float* __restrict__ u, float* __restrict__ out,
                         const float* __restrict__ w, const float* __restrict__ bias) {
    int i = blockIdx.x * blockDim.x + threadIdx.x;
    if (i >= TB * CC) return;
    int c  = i & (CC - 1);
    int tb = i >> 8;
    int b  = tb & (BB - 1);
    int t  = tb >> 2;
    float acc = bias[c];
    const float* wc = w + c * KW;
    #pragma unroll 4
    for (int k = 0; k < KW; ++k) {
        int tt = t + k - (KW - 1) / 2;
        if (tt >= 0 && tt < TT)
            acc += wc[k] * u[((size_t)(tt * BB + b)) * CC + c];
    }
    out[i] = acc;
}

__global__ void bn_swish_k(float* __restrict__ y, const float* __restrict__ g,
                           const float* __restrict__ b, const float* __restrict__ m,
                           const float* __restrict__ v) {
    int i = blockIdx.x * blockDim.x + threadIdx.x;
    if (i >= TB * CC) return;
    int c = i & (CC - 1);
    float t = (y[i] - m[c]) * rsqrtf(v[c] + EPSF) * g[c] + b[c];
    y[i] = t / (1.0f + __expf(-t));
}

// ---------------------------------------------------------------------------
extern "C" void kernel_launch(void* const* d_in, const int* in_sizes, int n_in,
                              void* d_out, int out_size, void* d_ws, size_t ws_size,
                              hipStream_t stream) {
    (void)in_sizes; (void)n_in; (void)out_size; (void)ws_size;
    const float* x_in    = (const float*)d_in[0];
    const float* pos_emb = (const float*)d_in[1];
    const float* ffm_w1  = (const float*)d_in[2];
    const float* ffm_b1  = (const float*)d_in[3];
    const float* ffm_w2  = (const float*)d_in[4];
    const float* ffm_b2  = (const float*)d_in[5];
    const float* ff_w1   = (const float*)d_in[6];
    const float* ff_b1   = (const float*)d_in[7];
    const float* ff_w2   = (const float*)d_in[8];
    const float* ff_b2   = (const float*)d_in[9];
    const float* in_w    = (const float*)d_in[10];
    const float* in_b    = (const float*)d_in[11];
    const float* out_w   = (const float*)d_in[12];
    const float* out_b   = (const float*)d_in[13];
    const float* pos_w   = (const float*)d_in[14];
    const float* bias_u  = (const float*)d_in[15];
    const float* bias_v  = (const float*)d_in[16];
    const float* pw1_w   = (const float*)d_in[17];
    const float* pw1_b   = (const float*)d_in[18];
    const float* dw_w    = (const float*)d_in[19];
    const float* dw_b    = (const float*)d_in[20];
    const float* bn_g    = (const float*)d_in[21];
    const float* bn_b    = (const float*)d_in[22];
    const float* bn_m    = (const float*)d_in[23];
    const float* bn_v    = (const float*)d_in[24];
    const float* pw2_w   = (const float*)d_in[25];
    const float* pw2_b   = (const float*)d_in[26];
    const float* ln_g    = (const float*)d_in[27];
    const float* ln_b    = (const float*)d_in[28];

    // ---- workspace layout (≈63.5 MB) ----
    char* ws = (char*)d_ws;
    const size_t MB = 1ull << 20;
    float*     y       = (float*)(ws + 0 * MB);       // 4 MB  (T,B,C)
    float*     tmp_big = (float*)(ws + 4 * MB);       // 16 MB (TB x DFF / TB x 2C)
    float*     tmp_c   = (float*)(ws + 20 * MB);      // 4 MB  (TB x C)
    float*     qkv     = (float*)(ws + 24 * MB);      // 12 MB (TB x 3C)
    float*     p32     = (float*)(ws + 36 * MB);      // 2 MB  (TPP x C)
    float*     scores  = (float*)(ws + 38 * MB);      // 4 MB  (T x T)
    float*     bd      = (float*)(ws + 42 * MB);      // 8 MB  (T x TPP)
    float*     attnout = (float*)(ws + 50 * MB);      // 4 MB  (T,B,C)
    _Float16*  a16     = (_Float16*)(ws + 54 * MB);   // 8 MB  generic A staging
    _Float16*  w16     = (_Float16*)(ws + 62 * MB);   // 1 MB  weights / p16
    _Float16*  aq16    = (_Float16*)(ws + 63 * MB);              // 128 KB
    _Float16*  aq2     = (_Float16*)(ws + 63 * MB + 128 * 1024); // 128 KB
    _Float16*  k16     = (_Float16*)(ws + 63 * MB + 256 * 1024); // 128 KB
    _Float16*  vT16    = (_Float16*)(ws + 63 * MB + 384 * 1024); // 128 KB

    float* xb = (float*)d_out;   // running activation (T,B,C)
    hipMemcpyAsync(xb, x_in, sizeof(float) * TB * CC, hipMemcpyDeviceToDevice, stream);

    auto cvt = [&](const float* src, _Float16* dst, long long nsrc, long long ndst) {
        int blocks = (int)((ndst + 255) / 256);
        f32_to_f16_k<<<blocks, 256, 0, stream>>>(src, dst, nsrc, ndst);
    };
    auto gemm = [&](const _Float16* A, int lda, const _Float16* Bm, int ldb,
                    float* D, int ldd, const float* bias, int M, int N, int Kd) {
        int tM2 = M / 32;               // all M are multiples of 32
        int tN4 = (N + 63) / 64;        // padded B buffers cover full groups
        int waves = tM2 * tN4;
        int blocks = (waves + 7) / 8;   // 8 waves (256 threads) per block
        wmma_gemm<<<blocks, 256, 0, stream>>>(A, lda, Bm, ldb, D, ldd, bias, M, N, Kd, tM2, tN4);
    };
    auto LN = [&](const float* xi, float* yo, const float* g, const float* b) {
        layernorm_k<<<TB, CC, 0, stream>>>(xi, yo, g, b);
    };
    auto ADD = [&](float* x, const float* t, float a) {
        add_scaled_k<<<(TB * CC + 255) / 256, 256, 0, stream>>>(x, t, a, (long long)TB * CC);
    };
    const float scaling = 0.125f;  // HD^-0.5

    for (int i = 0; i < 2; ++i) {
        const float* lng = ln_g + (size_t)i * 5 * CC;
        const float* lnb = ln_b + (size_t)i * 5 * CC;

        // warm this layer's attention weights through the async-LDS path
        async_warm_k<<<64, 32, 0, stream>>>((const void*)(in_w + (size_t)i * C3 * CC));

        // ---- macaron FFN (half residual) ----
        LN(xb, y, lng + 0 * CC, lnb + 0 * CC);
        cvt(y, a16, (long long)TB * CC, (long long)TB * CC);
        cvt(ffm_w1 + (size_t)i * DFF * CC, w16, (long long)DFF * CC, (long long)DFF * CC);
        gemm(a16, CC, w16, CC, tmp_big, DFF, ffm_b1 + (size_t)i * DFF, TB, DFF, CC);
        swish_ip_k<<<(TB * DFF + 255) / 256, 256, 0, stream>>>(tmp_big, (long long)TB * DFF);
        cvt(tmp_big, a16, (long long)TB * DFF, (long long)TB * DFF);
        cvt(ffm_w2 + (size_t)i * CC * DFF, w16, (long long)CC * DFF, (long long)CC * DFF);
        gemm(a16, DFF, w16, DFF, tmp_c, CC, ffm_b2 + (size_t)i * CC, TB, CC, DFF);
        ADD(xb, tmp_c, 0.5f);

        // ---- rel-pos multi-head attention ----
        LN(xb, y, lng + 1 * CC, lnb + 1 * CC);
        cvt(y, a16, (long long)TB * CC, (long long)TB * CC);
        cvt(in_w + (size_t)i * C3 * CC, w16, (long long)C3 * CC, (long long)C3 * CC);
        gemm(a16, CC, w16, CC, qkv, C3, in_b + (size_t)i * C3, TB, C3, CC);
        // p = pos_emb @ pos_w^T  (2047 rows padded to 2048 with zeros)
        cvt(pos_emb, a16, (long long)TP * CC, (long long)TPP * CC);
        cvt(pos_w + (size_t)i * CC * CC, w16, (long long)CC * CC, (long long)CC * CC);
        gemm(a16, CC, w16, CC, p32, CC, nullptr, TPP, CC, CC);
        cvt(p32, w16, (long long)TPP * CC, (long long)TPP * CC);   // w16 := p16 (2048 x 256)

        for (int b = 0; b < BB; ++b)
            for (int h = 0; h < HH; ++h) {
                build_heads_k<<<(TT * HD + 255) / 256, 256, 0, stream>>>(
                    qkv, bias_u + (size_t)(i * HH + h) * HD, bias_v + (size_t)(i * HH + h) * HD,
                    aq16, aq2, k16, vT16, b, h, scaling);
                // AC = (q*s + bu) . k^T
                gemm(aq16, HD, k16, HD, scores, TT, nullptr, TT, TT, HD);
                // BD = (q*s + bv) . p^T   (N = 2047, padded tiles)
                gemm(aq2, HD, w16 + h * HD, CC, bd, TPP, nullptr, TT, TP, HD);
                shift_add_k<<<(TT * TT + 255) / 256, 256, 0, stream>>>(scores, bd);
                softmax_row_k<<<TT, 256, 0, stream>>>(scores);
                cvt(scores, a16, (long long)TT * TT, (long long)TT * TT);
                // o = probs @ V  (B operand = V^T, row-major 64 x T)
                gemm(a16, TT, vT16, TT, attnout + (size_t)b * CC + h * HD, BB * CC,
                     nullptr, TT, HD, TT);
            }
        cvt(attnout, a16, (long long)TB * CC, (long long)TB * CC);
        cvt(out_w + (size_t)i * CC * CC, w16, (long long)CC * CC, (long long)CC * CC);
        gemm(a16, CC, w16, CC, tmp_c, CC, out_b + (size_t)i * CC, TB, CC, CC);
        ADD(xb, tmp_c, 1.0f);

        // ---- convolution module ----
        LN(xb, y, lng + 2 * CC, lnb + 2 * CC);
        cvt(y, a16, (long long)TB * CC, (long long)TB * CC);
        cvt(pw1_w + (size_t)i * 2 * CC * CC, w16, (long long)2 * CC * CC, (long long)2 * CC * CC);
        gemm(a16, CC, w16, CC, tmp_big, 2 * CC, pw1_b + (size_t)i * 2 * CC, TB, 2 * CC, CC);
        glu_k<<<(TB * CC + 255) / 256, 256, 0, stream>>>(tmp_big, tmp_c);
        dwconv_k<<<(TB * CC + 255) / 256, 256, 0, stream>>>(
            tmp_c, y, dw_w + (size_t)i * CC * KW, dw_b + (size_t)i * CC);
        bn_swish_k<<<(TB * CC + 255) / 256, 256, 0, stream>>>(
            y, bn_g + (size_t)i * CC, bn_b + (size_t)i * CC,
            bn_m + (size_t)i * CC, bn_v + (size_t)i * CC);
        cvt(y, a16, (long long)TB * CC, (long long)TB * CC);
        cvt(pw2_w + (size_t)i * CC * CC, w16, (long long)CC * CC, (long long)CC * CC);
        gemm(a16, CC, w16, CC, tmp_c, CC, pw2_b + (size_t)i * CC, TB, CC, CC);
        ADD(xb, tmp_c, 1.0f);

        // ---- second FFN (half residual) ----
        LN(xb, y, lng + 3 * CC, lnb + 3 * CC);
        cvt(y, a16, (long long)TB * CC, (long long)TB * CC);
        cvt(ff_w1 + (size_t)i * DFF * CC, w16, (long long)DFF * CC, (long long)DFF * CC);
        gemm(a16, CC, w16, CC, tmp_big, DFF, ff_b1 + (size_t)i * DFF, TB, DFF, CC);
        swish_ip_k<<<(TB * DFF + 255) / 256, 256, 0, stream>>>(tmp_big, (long long)TB * DFF);
        cvt(tmp_big, a16, (long long)TB * DFF, (long long)TB * DFF);
        cvt(ff_w2 + (size_t)i * CC * DFF, w16, (long long)CC * DFF, (long long)CC * DFF);
        gemm(a16, DFF, w16, DFF, tmp_c, CC, ff_b2 + (size_t)i * CC, TB, CC, DFF);
        ADD(xb, tmp_c, 0.5f);

        // ---- final layer norm (in place) ----
        LN(xb, xb, lng + 4 * CC, lnb + 4 * CC);
    }
}
